// HeatmapBatch_11330123727212
// MI455X (gfx1250) — matrix operand order repaired and checked by reading, not compiled, not run on hardware
//
#include <hip/hip_runtime.h>

typedef __attribute__((ext_vector_type(16))) _Float16 v16h;
typedef __attribute__((ext_vector_type(8)))  _Float16 v8h;
typedef __attribute__((ext_vector_type(4)))  _Float16 v4h;
typedef __attribute__((ext_vector_type(8)))  float    v8f;
typedef __attribute__((ext_vector_type(4)))  float    v4f;

namespace {
constexpr int IMGW = 256;          // H == W == 256
constexpr int NIMG = 64 * 21;      // B * K images
constexpr int BAND = 32;           // output rows per workgroup
constexpr int ROWS = 48;           // BAND + 16 (4-row halo each side, rounded to 16)
constexpr int INW  = 272;          // 4 left halo + 256 + 12 slack (keeps b128 reads in-bounds)
constexpr int QPR  = INW / 4;      // 68 quads per staged row
}

__global__ __launch_bounds__(256, 2)
void heatmap_sepconv_wmma(const int* __restrict__ coords,
                          const float* __restrict__ noise,
                          const float* __restrict__ k2d,
                          float* __restrict__ out)
{
  __shared__ alignas(16) _Float16 sIn[ROWS * INW];     // padded input band, f16
  __shared__ alignas(16) _Float16 sMidT[IMGW * ROWS];  // h-convolved, TRANSPOSED [col][row]
  __shared__ alignas(16) _Float16 sW[16 * 32];         // banded taps: sW[i][j] = w1d[j-i]

  const int img   = blockIdx.y;          // 0..1343  (b*K + k)
  const int band  = blockIdx.x;          // 0..7
  const int ybase = band * BAND;
  const int r0    = ybase - 4;           // first image row staged (may be <0 -> zero pad)
  const int tid   = threadIdx.x;
  const int lane  = tid & 31;
  const int wave  = tid >> 5;
  const int n16   = lane & 15;
  const bool hihalf = (lane & 16) != 0;

  // ---- separable taps: kernel2d = outer(g,g); w1d[j] = k2d[4][j]/sqrt(k2d[4][4]) ----
  const float kc  = k2d[40];             // center == max of 2D kernel
  const float inv = rsqrtf(kc);

  // ---- build the 16x32 banded-weight table once (2 entries per thread) ----
  #pragma unroll
  for (int e = 0; e < 2; ++e) {
    const int idx = tid + e * 256;       // 0..511
    const int i = idx >> 5, j = idx & 31;
    const int d = j - i;
    float wv = 0.0f;
    if (d >= 0 && d <= 8) wv = k2d[36 + d] * inv;
    sW[idx] = (_Float16)wv;
  }

  // ---- stage scaled+padded input band into LDS as f16, 4 elements at a time ----
  // quad q covers padded cols 4q..4q+3 (image cols 4q-4..4q-1): fully in iff 1<=q<=64
  const float* nimg = noise + (size_t)img * IMGW * IMGW;
  for (int idx = tid; idx < ROWS * QPR; idx += 256) {
    const int r = idx / QPR;
    const int q = idx - r * QPR;
    const int y = r0 + r;
    v4h hv = {};
    if ((unsigned)y < (unsigned)IMGW && q >= 1 && q <= 64) {
      const v4f vv = __builtin_nontemporal_load((const v4f*)&nimg[y * IMGW + (q << 2) - 4]);
      #pragma unroll
      for (int i = 0; i < 4; ++i) hv[i] = (_Float16)(0.01f * vv[i]);
    }
    *(v4h*)&sIn[r * INW + (q << 2)] = hv;
  }
  __syncthreads();
  if (tid == 0) {                        // insert the keypoint spike (value 1/max(kernel2d))
    const int sx = coords[2 * img + 0];
    const int sy = coords[2 * img + 1];
    const int r  = sy - r0;
    if (0 <= r && r < ROWS) sIn[r * INW + sx + 4] = (_Float16)(1.0f / kc);
  }
  __syncthreads();

  // ---- materialize constant WMMA operands from the shared table ----
  // Horizontal B[k][n] = w[k-n]: lanes0-15 N=lane K={0..7,16..23}; lanes16-31 N=lane-16 K={8..15,24..31}
  v16h bH;
  {
    const int o = n16 * 32 + (hihalf ? 8 : 0);
    const v8h lo8 = *(const v8h*)&sW[o];
    const v8h hi8 = *(const v8h*)&sW[o + 16];
    #pragma unroll
    for (int i = 0; i < 8; ++i) { bH[i] = lo8[i]; bH[8 + i] = hi8[i]; }
  }
  // Vertical A[m][k] = w[k-m]: lanes0-15 M=lane K=0..15; lanes16-31 K=16..31
  v16h aV;
  {
    const int o = n16 * 32 + (hihalf ? 16 : 0);
    const v8h lo8 = *(const v8h*)&sW[o];
    const v8h hi8 = *(const v8h*)&sW[o + 8];
    #pragma unroll
    for (int i = 0; i < 8; ++i) { aV[i] = lo8[i]; aV[8 + i] = hi8[i]; }
  }

  // ---- pass 1: horizontal 9-tap conv, one WMMA per 16x16 tile (48 tiles / block) ----
  for (int t = wave; t < (ROWS / 16) * 16; t += 8) {
    const int rg  = t >> 4;              // row group 0..2
    const int x0  = (t & 15) << 4;       // tile col base 0..240
    const int row = (rg << 4) + n16;
    const int cb  = x0 + (hihalf ? 8 : 0);
    const v8h lo8 = *(const v8h*)&sIn[row * INW + cb];        // K 0-7 / 8-15
    const v8h hi8 = *(const v8h*)&sIn[row * INW + cb + 16];   // K 16-23 / 24-31 (zero-weighted slack)
    v16h a;
    #pragma unroll
    for (int i = 0; i < 8; ++i) { a[i] = lo8[i]; a[8 + i] = hi8[i]; }
    v8f c = {};
    c = __builtin_amdgcn_wmma_f32_16x16x32_f16(false, a, false, bH, (short)0, c, false, false);
    // D layout: VGPR i = row (rg*16 + i + 8*hihalf), col x0+n16 -> transposed store is contiguous
    v8h m8;
    #pragma unroll
    for (int i = 0; i < 8; ++i) m8[i] = (_Float16)c[i];
    const int col = x0 + n16;
    const int rb  = (rg << 4) + (hihalf ? 8 : 0);
    *(v8h*)&sMidT[col * ROWS + rb] = m8;
  }
  __syncthreads();

  // ---- pass 2: vertical 9-tap conv + ReLU + NT store (32 tiles / block) ----
  for (int t = wave; t < (BAND / 16) * 16; t += 8) {
    const int g   = t >> 4;              // 0..1
    const int x0  = (t & 15) << 4;
    const int col = x0 + n16;
    const int rb  = (g << 4) + (hihalf ? 16 : 0);   // B: lanes0-15 K=0..15, lanes16-31 K=16..31
    const v8h lo8 = *(const v8h*)&sMidT[col * ROWS + rb];
    const v8h hi8 = *(const v8h*)&sMidT[col * ROWS + rb + 8];
    v16h b;
    #pragma unroll
    for (int i = 0; i < 8; ++i) { b[i] = lo8[i]; b[8 + i] = hi8[i]; }
    v8f c = {};
    c = __builtin_amdgcn_wmma_f32_16x16x32_f16(false, aV, false, b, (short)0, c, false, false);
    const int y0 = ybase + (g << 4) + (hihalf ? 8 : 0);
    float* op = out + (size_t)img * IMGW * IMGW + (size_t)y0 * IMGW + x0 + n16;
    #pragma unroll
    for (int i = 0; i < 8; ++i)
      __builtin_nontemporal_store(fmaxf(c[i], 0.0f), &op[i * IMGW]);
  }
}

extern "C" void kernel_launch(void* const* d_in, const int* in_sizes, int n_in,
                              void* d_out, int out_size, void* d_ws, size_t ws_size,
                              hipStream_t stream) {
  (void)in_sizes; (void)n_in; (void)d_ws; (void)ws_size; (void)out_size;
  const int*   coords = (const int*)d_in[0];   // (B,K,2) int32, [...,(x,y)]
  const float* noise  = (const float*)d_in[1]; // (B,K,256,256) f32
  const float* k2d    = (const float*)d_in[2]; // (9,9) f32
  float* out = (float*)d_out;                  // (B,K,256,256) f32
  dim3 grid(IMGW / BAND, NIMG);                // 8 bands x 1344 images
  heatmap_sepconv_wmma<<<grid, 256, 0, stream>>>(coords, noise, k2d, out);
}
